// Bert_40063454937521
// MI455X (gfx1250) — matrix operand order repaired and checked by reading, not compiled, not run on hardware
//
#include <hip/hip_runtime.h>
#include <math.h>

// ---------------------------------------------------------------------------
// DeBERTa-style encoder forward for MI455X (gfx1250).
// - All contractions on v_wmma_f32_16x16x32_bf16 (wave32, f32 accum).
// - Dense GEMMs double-buffer their LDS tiles with GLOBAL_LOAD_ASYNC_TO_LDS_B128
//   (ASYNCcnt-tracked), overlapping the next tile's fetch with WMMA compute.
// ---------------------------------------------------------------------------

typedef unsigned short u16;
typedef unsigned int   u32;
typedef __attribute__((ext_vector_type(8)))  u16    u16x8;
typedef __attribute__((ext_vector_type(16))) __bf16 v16bf;
typedef __attribute__((ext_vector_type(8)))  float  v8f;

#define S_LEN  512
#define B_SZ   8
#define H_DIM  768
#define NHEAD  12
#define D_HEAD 64
#define LAYERS 12
#define I_DIM  2048
#define NBUCK  63          // 2*BK-1
#define ROWS   4096        // S*B
#define H2     1536        // 2*H
#define I2     4096        // 2*I
#define ATT_SCALE 0.07216878364870323f   // 1/sqrt(3*64)

__device__ __forceinline__ u16 f2bf(float f) {
  u32 u = __builtin_bit_cast(u32, f);
  u32 r = (u + 0x7FFFu + ((u >> 16) & 1u)) >> 16;   // round-to-nearest-even
  return (u16)r;
}

__device__ __forceinline__ v8f wmma_bf16(v16bf a, v16bf b, v8f c) {
  // 8 args: (neg_a, A, neg_b, B, c_mod, C, reuse_a, reuse_b)
  return __builtin_amdgcn_wmma_f32_16x16x32_bf16(false, a, false, b, (short)0, c,
                                                 false, false);
}

// Async global -> LDS copy of 16 bytes (per-lane addresses), ASYNCcnt-tracked.
// Low 32 bits of a generic LDS pointer are the wave-relative LDS byte address
// (ISA 10.2 aperture rule), which is exactly what VDST wants.
__device__ __forceinline__ void async_b128(void* lds_dst, const void* gsrc) {
  unsigned ldsa = (unsigned)(unsigned long long)lds_dst;
  asm volatile("global_load_async_to_lds_b128 %0, %1, off"
               :: "v"(ldsa), "v"(gsrc) : "memory");
}

union FragU { u16x8 h[2]; v16bf v; u16 u[16]; };

// A fragment 16x32 bf16 from row-major source (ld elems/row).
// lane L<16: row=L, K 0..15; lane L>=16: row=L-16, K 16..31.
__device__ __forceinline__ v16bf frag_a(const u16* p, long ld) {
  int lane = threadIdx.x & 31;
  int r = lane & 15, half = lane >> 4;
  const u16* q = p + (long)r * ld + half * 16;
  FragU f;
  f.h[0] = *(const u16x8*)(q);
  f.h[1] = *(const u16x8*)(q + 8);
  return f.v;
}

// B fragment 32x16: source stored N rows x K contiguous (ld elems/N-row).
// lane L<16: col=L, K 0..15; lane L>=16: col=L-16, K 16..31.
__device__ __forceinline__ v16bf frag_b(const u16* p, long ld) {
  int lane = threadIdx.x & 31;
  int n = lane & 15, half = lane >> 4;
  const u16* q = p + (long)n * ld + half * 16;
  FragU f;
  f.h[0] = *(const u16x8*)(q);
  f.h[1] = *(const u16x8*)(q + 8);
  return f.v;
}

// B fragment from K-major source (row = k, ld elems per k-row).
__device__ __forceinline__ v16bf frag_b_kmajor(const u16* p, long ld) {
  int lane = threadIdx.x & 31;
  int n = lane & 15;
  long half = (long)(lane >> 4) * 16;
  FragU f;
  #pragma unroll
  for (int e = 0; e < 16; ++e) f.u[e] = p[(half + e) * ld + n];
  return f.v;
}

// ---------------------------------------------------------------------------
// Tiled GEMM: out[M,N] = A[M,K](bf16) x Wt[N,K](bf16)^T (+bias) (+residual)
// Block: 256 thr = 8 waves; tile 128M x 64N; each wave 32x32 (2x2 WMMA accs).
// Double-buffered async global->LDS staging (3 async b128 per thread per tile).
// ---------------------------------------------------------------------------
template <int MODE>   // 0: out = acc(+bias); 1: out = res + acc(+bias)
__global__ __launch_bounds__(256) void gemm_kernel(
    const u16* __restrict__ A, const u16* __restrict__ Wt,
    const float* __restrict__ bias, const float* __restrict__ res,
    float* __restrict__ out, int M, int N, int K)
{
  const int LDT = 48;                       // padded LDS row (16B-aligned rows)
  const int TILE = (128 + 64) * LDT;        // u16 elems per buffer
  extern __shared__ char smem[];
  u16* buf0 = (u16*)smem;
  u16* buf1 = buf0 + TILE;
  const int m0 = blockIdx.y * 128, n0 = blockIdx.x * 64;
  const int t = threadIdx.x;
  const int wave = t >> 5, lane = t & 31;
  const int wm = (wave & 3) * 32;           // 4 waves along M
  const int wn = (wave >> 2) * 32;          // 2 waves along N

  // Per-thread staging coordinates (row clamped: garbage rows are never stored)
  const int ar0 = (t + 0) >> 2,   ac0 = ((t + 0) & 3) * 8;
  const int ar1 = (t + 256) >> 2, ac1 = ((t + 256) & 3) * 8;
  const int wr  = t >> 2,         wc  = (t & 3) * 8;
  const int gra0 = (m0 + ar0 < M) ? m0 + ar0 : M - 1;
  const int gra1 = (m0 + ar1 < M) ? m0 + ar1 : M - 1;
  const int grw  = (n0 + wr  < N) ? n0 + wr  : N - 1;
  const long ga0 = (long)gra0 * K + ac0;
  const long ga1 = (long)gra1 * K + ac1;
  const long gw  = (long)grw  * K + wc;

  auto issue = [&](u16* buf, int k0) {
    async_b128(buf + ar0 * LDT + ac0, A + ga0 + k0);
    async_b128(buf + ar1 * LDT + ac1, A + ga1 + k0);
    async_b128(buf + 128 * LDT + wr * LDT + wc, Wt + gw + k0);
  };

  v8f acc[2][2] = {};
  issue(buf0, 0);
  u16* cur = buf0;
  u16* nxt = buf1;
  for (int k0 = 0; k0 < K; k0 += 32) {
    if (k0 + 32 < K) {
      issue(nxt, k0 + 32);                          // 6 in flight
      asm volatile("s_wait_asynccnt 0x3" ::: "memory");   // oldest 3 retired
    } else {
      asm volatile("s_wait_asynccnt 0x0" ::: "memory");
    }
    __syncthreads();
    u16* As = cur;
    u16* Ws = cur + 128 * LDT;
    v16bf a0 = frag_a(As + (long)(wm +  0) * LDT, LDT);
    v16bf a1 = frag_a(As + (long)(wm + 16) * LDT, LDT);
    v16bf b0 = frag_b(Ws + (long)(wn +  0) * LDT, LDT);
    v16bf b1 = frag_b(Ws + (long)(wn + 16) * LDT, LDT);
    acc[0][0] = wmma_bf16(a0, b0, acc[0][0]);
    acc[0][1] = wmma_bf16(a0, b1, acc[0][1]);
    acc[1][0] = wmma_bf16(a1, b0, acc[1][0]);
    acc[1][1] = wmma_bf16(a1, b1, acc[1][1]);
    __syncthreads();                                // readers done before reuse
    u16* tmpp = cur; cur = nxt; nxt = tmpp;
  }

  const int cn = lane & 15, cmb = (lane >> 4) * 8;
  #pragma unroll
  for (int i = 0; i < 2; ++i)
    #pragma unroll
    for (int j = 0; j < 2; ++j) {
      int gn = n0 + wn + j * 16 + cn;
      #pragma unroll
      for (int r = 0; r < 8; ++r) {
        int gm = m0 + wm + i * 16 + cmb + r;
        if (gm < M) {
          float v = acc[i][j][r];
          if (bias) v += bias[gn];
          if (MODE == 1) v += res[(long)gm * N + gn];
          out[(long)gm * N + gn] = v;
        }
      }
    }
}

// ---------------------------------------------------------------------------
// Positional score GEMMs: CQ[bh][q][r] = q . kpos_r ; CK[bh][k][r] = k . qpos_r
// grid (B*NH, 2): y=0 -> CQ, y=1 -> CK.  [512 x 64] out, K = 64.
// ---------------------------------------------------------------------------
__global__ __launch_bounds__(256) void pos_scores_kernel(
    const u16* __restrict__ QKb, const u16* __restrict__ POSb,
    float* __restrict__ CQ, float* __restrict__ CK)
{
  int bh = blockIdx.x;
  int sel = blockIdx.y;                       // 0: q x kpos; 1: k x qpos
  int b = bh / NHEAD, h = bh % NHEAD;
  const long ldq = (long)B_SZ * H2;           // 12288
  const u16* A  = QKb + (long)b * H2 + (sel ? H_DIM : 0) + h * D_HEAD;
  const u16* Bm = POSb + (sel ? 0 : H_DIM) + h * D_HEAD;
  float* Out = (sel ? CK : CQ) + (long)bh * S_LEN * 64;
  int lane = threadIdx.x & 31, wave = threadIdx.x >> 5;

  v16bf bf[4][2];
  #pragma unroll
  for (int nt = 0; nt < 4; ++nt)
    #pragma unroll
    for (int ks = 0; ks < 2; ++ks)
      bf[nt][ks] = frag_b(Bm + (long)(nt * 16) * H2 + ks * 32, H2);

  int cn = lane & 15, cmb = (lane >> 4) * 8;
  for (int mt = wave; mt < 32; mt += 8) {
    v16bf a[2];
    #pragma unroll
    for (int ks = 0; ks < 2; ++ks)
      a[ks] = frag_a(A + (long)(mt * 16) * ldq + ks * 32, ldq);
    #pragma unroll
    for (int nt = 0; nt < 4; ++nt) {
      v8f c = {};
      c = wmma_bf16(a[0], bf[nt][0], c);
      c = wmma_bf16(a[1], bf[nt][1], c);
      #pragma unroll
      for (int r = 0; r < 8; ++r)
        Out[(long)(mt * 16 + cmb + r) * 64 + nt * 16 + cn] = c[r];
    }
  }
}

// ---------------------------------------------------------------------------
// Fused attention: one block per (b, h, 64-query tile).
// phase1: scores = q k^T (WMMA) + bucket-gathered pos terms, mask -> LDS
// phase2: row softmax in LDS, p -> bf16 LDS
// phase3: ctx = p v (WMMA) -> CTX
// Dynamic LDS: 64*512*4 + 64*512*2 + 512*4 = 198656 B (within 320KB/WGP).
// ---------------------------------------------------------------------------
__global__ __launch_bounds__(256) void attn_kernel(
    const u16* __restrict__ QKb, const u16* __restrict__ Vb,
    const float* __restrict__ CQ, const float* __restrict__ CK,
    const int* __restrict__ PIDX, const unsigned char* __restrict__ mask,
    float* __restrict__ CTX)
{
  extern __shared__ char smem[];
  float* sc  = (float*)smem;                               // 64 x 512 f32
  u16*   pb  = (u16*)(smem + 64 * 512 * 4);                // 64 x 512 bf16
  float* rmax = (float*)(smem + 64 * 512 * 4 + 64 * 512 * 2);
  float* rsum = rmax + 256;

  const int q0 = blockIdx.x * 64;
  const int h  = blockIdx.y;
  const int b  = blockIdx.z;
  const int bh = b * NHEAD + h;
  const int t = threadIdx.x, lane = t & 31, wave = t >> 5;
  const long ldq = (long)B_SZ * H2;
  const u16* Aq = QKb + (long)b * H2 + h * D_HEAD;         // + s*ldq
  const u16* Ak = Aq + H_DIM;

  { // phase 1 ---------------------------------------------------------------
    int mt  = wave & 3;                                    // 16-row q tile
    int nh2 = wave >> 2;                                   // 256-col half
    v16bf aq[2];
    #pragma unroll
    for (int ks = 0; ks < 2; ++ks)
      aq[ks] = frag_a(Aq + (long)(q0 + mt * 16) * ldq + ks * 32, ldq);
    int cn = lane & 15, cmb = (lane >> 4) * 8;
    for (int j = 0; j < 16; ++j) {
      int nt = nh2 * 16 + j;
      v8f c = {};
      #pragma unroll
      for (int ks = 0; ks < 2; ++ks) {
        v16bf bk = frag_b(Ak + (long)(nt * 16) * ldq + ks * 32, ldq);
        c = wmma_bf16(aq[ks], bk, c);
      }
      int kcol = nt * 16 + cn;
      bool mk = mask[b * S_LEN + kcol] != 0;
      const float* ckrow = CK + ((long)bh * S_LEN + kcol) * 64;
      #pragma unroll
      for (int r = 0; r < 8; ++r) {
        int ql = mt * 16 + cmb + r;
        int qg = q0 + ql;
        int idx = PIDX[qg * S_LEN + kcol];
        float v = (c[r] + CQ[((long)bh * S_LEN + qg) * 64 + idx] + ckrow[idx]) *
                  ATT_SCALE;
        if (mk) v = -1e9f;
        sc[ql * 512 + kcol] = v;
      }
    }
  }
  __syncthreads();
  { // phase 2: softmax ------------------------------------------------------
    int row = t >> 2, qtr = t & 3;
    float mx = -3.0e38f;
    for (int k = qtr * 128; k < qtr * 128 + 128; ++k)
      mx = fmaxf(mx, sc[row * 512 + k]);
    rmax[row * 4 + qtr] = mx;
    __syncthreads();
    mx = fmaxf(fmaxf(rmax[row * 4 + 0], rmax[row * 4 + 1]),
               fmaxf(rmax[row * 4 + 2], rmax[row * 4 + 3]));
    float sm = 0.f;
    for (int k = qtr * 128; k < qtr * 128 + 128; ++k)
      sm += __expf(sc[row * 512 + k] - mx);
    rsum[row * 4 + qtr] = sm;
    __syncthreads();
    sm = rsum[row * 4 + 0] + rsum[row * 4 + 1] + rsum[row * 4 + 2] +
         rsum[row * 4 + 3];
    float inv = 1.0f / sm;
    for (int k = qtr * 128; k < qtr * 128 + 128; ++k) {
      float p = __expf(sc[row * 512 + k] - mx) * inv;
      if (mask[b * S_LEN + k]) p = 0.f;
      pb[row * 512 + k] = f2bf(p);
    }
  }
  __syncthreads();
  { // phase 3: ctx = p @ v --------------------------------------------------
    int mt = wave >> 1;
    int npair = wave & 1;
    v8f cc[2] = {};
    for (int ks = 0; ks < 16; ++ks) {
      v16bf ap = frag_a(pb + (long)(mt * 16) * 512 + ks * 32, 512);
      #pragma unroll
      for (int jj = 0; jj < 2; ++jj) {
        int nt = npair * 2 + jj;
        const u16* vb = Vb + ((long)(ks * 32) * B_SZ + b) * H_DIM + h * D_HEAD +
                        nt * 16;
        v16bf bv = frag_b_kmajor(vb, (long)B_SZ * H_DIM);
        cc[jj] = wmma_bf16(ap, bv, cc[jj]);
      }
    }
    int cn = lane & 15, cmb = (lane >> 4) * 8;
    #pragma unroll
    for (int jj = 0; jj < 2; ++jj) {
      int nt = npair * 2 + jj;
      #pragma unroll
      for (int r = 0; r < 8; ++r) {
        int qg = q0 + mt * 16 + cmb + r;
        CTX[((long)qg * B_SZ + b) * H_DIM + h * D_HEAD + nt * 16 + cn] = cc[jj][r];
      }
    }
  }
}

// ---------------------------------------------------------------------------
// LayerNorm / elementwise kernels (one row per block, 256 threads)
// ---------------------------------------------------------------------------
__global__ __launch_bounds__(256) void ln_kernel(
    const float* __restrict__ in, float* __restrict__ outF, u16* __restrict__ outB,
    const float* __restrict__ gamma, const float* __restrict__ beta, int cols)
{
  __shared__ float red[256], red2[256];
  long row = blockIdx.x;
  const float* x = in + row * cols;
  float s = 0.f, s2 = 0.f;
  for (int c = threadIdx.x; c < cols; c += 256) { float v = x[c]; s += v; s2 += v * v; }
  red[threadIdx.x] = s; red2[threadIdx.x] = s2;
  __syncthreads();
  for (int o = 128; o > 0; o >>= 1) {
    if ((int)threadIdx.x < o) {
      red[threadIdx.x] += red[threadIdx.x + o];
      red2[threadIdx.x] += red2[threadIdx.x + o];
    }
    __syncthreads();
  }
  float mean = red[0] / cols;
  float var  = red2[0] / cols - mean * mean;
  float inv  = rsqrtf(var + 1e-7f);
  for (int c = threadIdx.x; c < cols; c += 256) {
    float v = (x[c] - mean) * inv;
    if (gamma) v = v * gamma[c] + beta[c];
    if (outF) outF[row * cols + c] = v;
    if (outB) outB[row * cols + c] = f2bf(v);
  }
}

__global__ __launch_bounds__(256) void embed_ln_kernel(
    const int* __restrict__ ids, const float* __restrict__ emb, float* __restrict__ X)
{
  __shared__ float red[256], red2[256];
  long row = blockIdx.x;
  const float* x = emb + (long)ids[row] * H_DIM;
  float s = 0.f, s2 = 0.f;
  for (int c = threadIdx.x; c < H_DIM; c += 256) { float v = x[c]; s += v; s2 += v * v; }
  red[threadIdx.x] = s; red2[threadIdx.x] = s2;
  __syncthreads();
  for (int o = 128; o > 0; o >>= 1) {
    if ((int)threadIdx.x < o) {
      red[threadIdx.x] += red[threadIdx.x + o];
      red2[threadIdx.x] += red2[threadIdx.x + o];
    }
    __syncthreads();
  }
  float mean = red[0] / H_DIM;
  float inv  = rsqrtf(red2[0] / H_DIM - mean * mean + 1e-7f);
  for (int c = threadIdx.x; c < H_DIM; c += 256)
    X[row * H_DIM + c] = (x[c] - mean) * inv;
}

__global__ __launch_bounds__(256) void gate_ln_kernel(
    const float* __restrict__ CTX, const float* __restrict__ VG, u16* __restrict__ CTXb)
{
  __shared__ float tb[H_DIM];
  __shared__ float red[256], red2[256];
  long row = blockIdx.x;
  float s = 0.f, s2 = 0.f;
  for (int c = threadIdx.x; c < H_DIM; c += 256) {
    float g  = VG[row * H2 + H_DIM + c];
    float ge = 0.5f * g * (1.0f + erff(g * 0.70710678118f));   // exact GELU
    float v  = CTX[row * H_DIM + c] * ge;
    tb[c] = v; s += v; s2 += v * v;
  }
  red[threadIdx.x] = s; red2[threadIdx.x] = s2;
  __syncthreads();
  for (int o = 128; o > 0; o >>= 1) {
    if ((int)threadIdx.x < o) {
      red[threadIdx.x] += red[threadIdx.x + o];
      red2[threadIdx.x] += red2[threadIdx.x + o];
    }
    __syncthreads();
  }
  float mean = red[0] / H_DIM;
  float inv  = rsqrtf(red2[0] / H_DIM - mean * mean + 1e-7f);
  for (int c = threadIdx.x; c < H_DIM; c += 256)
    CTXb[row * H_DIM + c] = f2bf((tb[c] - mean) * inv);
}

__global__ __launch_bounds__(256) void geglu_ln_kernel(
    const float* __restrict__ FF, u16* __restrict__ FFb)
{
  __shared__ float tb[I_DIM];
  __shared__ float red[256], red2[256];
  long row = blockIdx.x;
  float s = 0.f, s2 = 0.f;
  for (int c = threadIdx.x; c < I_DIM; c += 256) {
    float a  = FF[row * I2 + c];
    float g  = FF[row * I2 + I_DIM + c];
    float ge = 0.5f * g * (1.0f + tanhf(0.7978845608f * (g + 0.044715f * g * g * g)));
    float v  = a * ge;
    tb[c] = v; s += v; s2 += v * v;
  }
  red[threadIdx.x] = s; red2[threadIdx.x] = s2;
  __syncthreads();
  for (int o = 128; o > 0; o >>= 1) {
    if ((int)threadIdx.x < o) {
      red[threadIdx.x] += red[threadIdx.x + o];
      red2[threadIdx.x] += red2[threadIdx.x + o];
    }
    __syncthreads();
  }
  float mean = red[0] / I_DIM;
  float inv  = rsqrtf(red2[0] / I_DIM - mean * mean + 1e-7f);
  for (int c = threadIdx.x; c < I_DIM; c += 256)
    FFb[row * I_DIM + c] = f2bf((tb[c] - mean) * inv);
}

// ---------------------------------------------------------------------------
// Small helpers
// ---------------------------------------------------------------------------
__global__ void bucket_kernel(int* __restrict__ PIDX) {
  int i = blockIdx.x * 256 + threadIdx.x;
  if (i >= S_LEN * S_LEN) return;
  int q = i >> 9, k = i & 511;
  int rel = q - k;
  int sgn = (rel > 0) - (rel < 0);
  int a = rel < 0 ? -rel : rel;
  int apos = (rel < 16 && rel > -16) ? 15 : (a < 511 ? a : 511);
  int bucket;
  if (apos <= 16) bucket = rel;
  else {
    float lp = ceilf(logf((float)apos / 16.0f) / logf(511.0f / 16.0f) * 15.0f) + 16.0f;
    bucket = (int)lp * sgn;
  }
  PIDX[i] = bucket + 31;                   // in [0, 62]
}

__global__ void convert_bf16_kernel(const float* __restrict__ src,
                                    u16* __restrict__ dst, long n) {
  long i = (long)blockIdx.x * 256 + threadIdx.x;
  if (i < n) dst[i] = f2bf(src[i]);
}

__global__ void convert_rows_kernel(const float* __restrict__ src,
                                    u16* __restrict__ dst, int src_ld, int cols) {
  long row = blockIdx.x;
  for (int c = threadIdx.x; c < cols; c += 256)
    dst[row * cols + c] = f2bf(src[row * src_ld + c]);
}

__global__ void zero_f32_kernel(float* __restrict__ p, int n) {
  int i = blockIdx.x * 256 + threadIdx.x;
  if (i < n) p[i] = 0.f;
}

// ---------------------------------------------------------------------------
// Host orchestration
// ---------------------------------------------------------------------------
extern "C" void kernel_launch(void* const* d_in, const int* in_sizes, int n_in,
                              void* d_out, int out_size, void* d_ws, size_t ws_size,
                              hipStream_t stream)
{
  (void)in_sizes; (void)n_in; (void)out_size; (void)ws_size;
  const int*           ids   = (const int*)d_in[0];
  const unsigned char* amask = (const unsigned char*)d_in[1];   // [B,1,1,S] bool
  const float* wemb = (const float*)d_in[2];
  const float* remb = (const float*)d_in[3];
  const float* relg = (const float*)d_in[4];
  const float* relbb= (const float*)d_in[5];
  const float* Wqk  = (const float*)d_in[6];
  const float* bqk  = (const float*)d_in[7];
  const float* Wvg  = (const float*)d_in[8];
  const float* bvg  = (const float*)d_in[9];
  const float* Wo   = (const float*)d_in[10];
  const float* bo   = (const float*)d_in[11];
  const float* W1   = (const float*)d_in[12];
  const float* W2   = (const float*)d_in[13];

  size_t off = 0;
  auto alloc = [&](size_t bytes) -> char* {
    char* p = (char*)d_ws + off;
    off += (bytes + 255) & ~(size_t)255;
    return p;
  };
  float* X    = (float*)alloc((size_t)ROWS * H_DIM * 4);
  u16*   HS   = (u16*)  alloc((size_t)ROWS * H_DIM * 2);
  float* QK   = (float*)alloc((size_t)ROWS * H2 * 4);
  float* VG   = (float*)alloc((size_t)ROWS * H2 * 4);
  u16*   QKb  = (u16*)  alloc((size_t)ROWS * H2 * 2);
  u16*   Vb   = (u16*)  alloc((size_t)ROWS * H_DIM * 2);
  u16*   RELb = (u16*)  alloc((size_t)64 * H_DIM * 2);
  float* POS  = (float*)alloc((size_t)64 * H2 * 4);
  u16*   POSb = (u16*)  alloc((size_t)64 * H2 * 2);
  float* CQ   = (float*)alloc((size_t)96 * S_LEN * 64 * 4);
  float* CK   = (float*)alloc((size_t)96 * S_LEN * 64 * 4);
  int*   PIDX = (int*)  alloc((size_t)S_LEN * S_LEN * 4);
  float* CTX  = (float*)alloc((size_t)ROWS * H_DIM * 4);
  u16*   CTXb = (u16*)  alloc((size_t)ROWS * H_DIM * 2);
  float* FF   = (float*)alloc((size_t)ROWS * I2 * 4);
  u16*   FFb  = (u16*)  alloc((size_t)ROWS * I_DIM * 2);
  u16*   WSTG = (u16*)  alloc((size_t)I2 * H_DIM * 2);    // weight staging (bf16)

  const size_t GEMM_LDS = (size_t)2 * (128 + 64) * 48 * 2;             // 36864 B
  const size_t ATT_LDS  = (size_t)64 * 512 * 4 + 64 * 512 * 2 + 512 * 4;

  bucket_kernel<<<(S_LEN * S_LEN + 255) / 256, 256, 0, stream>>>(PIDX);
  embed_ln_kernel<<<ROWS, 256, 0, stream>>>(ids, wemb, X);
  ln_kernel<<<NBUCK, 256, 0, stream>>>(remb, nullptr, RELb, relg, relbb, H_DIM);

  for (int i = 0; i < LAYERS; ++i) {
    const float* Wqk_i = Wqk + (size_t)i * H2 * H_DIM;
    const float* bqk_i = bqk + (size_t)i * H2;
    const float* Wvg_i = Wvg + (size_t)i * H2 * H_DIM;
    const float* bvg_i = bvg + (size_t)i * H2;
    const float* Wo_i  = Wo  + (size_t)i * H_DIM * H_DIM;
    const float* bo_i  = bo  + (size_t)i * H_DIM;
    const float* W1_i  = W1  + (size_t)i * I2 * H_DIM;
    const float* W2_i  = W2  + (size_t)i * H_DIM * I_DIM;

    // hs = LN(x) (bf16)
    ln_kernel<<<ROWS, 256, 0, stream>>>(X, nullptr, HS, nullptr, nullptr, H_DIM);

    // qk = hs Wqk^T + bqk ; pos = rel Wqk^T + bqk
    convert_bf16_kernel<<<((long)H2 * H_DIM + 255) / 256, 256, 0, stream>>>(
        Wqk_i, WSTG, (long)H2 * H_DIM);
    gemm_kernel<0><<<dim3(H2 / 64, ROWS / 128), 256, GEMM_LDS, stream>>>(
        HS, WSTG, bqk_i, nullptr, QK, ROWS, H2, H_DIM);
    gemm_kernel<0><<<dim3(H2 / 64, 1), 256, GEMM_LDS, stream>>>(
        RELb, WSTG, bqk_i, nullptr, POS, NBUCK, H2, H_DIM);
    zero_f32_kernel<<<(H2 + 255) / 256, 256, 0, stream>>>(POS + (size_t)NBUCK * H2, H2);
    convert_bf16_kernel<<<((long)64 * H2 + 255) / 256, 256, 0, stream>>>(
        POS, POSb, (long)64 * H2);

    // vg = hs Wvg^T + bvg
    convert_bf16_kernel<<<((long)H2 * H_DIM + 255) / 256, 256, 0, stream>>>(
        Wvg_i, WSTG, (long)H2 * H_DIM);
    gemm_kernel<0><<<dim3(H2 / 64, ROWS / 128), 256, GEMM_LDS, stream>>>(
        HS, WSTG, bvg_i, nullptr, VG, ROWS, H2, H_DIM);

    convert_bf16_kernel<<<((long)ROWS * H2 + 255) / 256, 256, 0, stream>>>(
        QK, QKb, (long)ROWS * H2);
    convert_rows_kernel<<<ROWS, 256, 0, stream>>>(VG, Vb, H2, H_DIM);

    // positional score tables + fused attention
    pos_scores_kernel<<<dim3(B_SZ * NHEAD, 2), 256, 0, stream>>>(QKb, POSb, CQ, CK);
    attn_kernel<<<dim3(S_LEN / 64, NHEAD, B_SZ), 256, ATT_LDS, stream>>>(
        QKb, Vb, CQ, CK, PIDX, amask, CTX);

    // ctx = LN(ctx * gelu(g)); x += ctx Wout^T + bout
    gate_ln_kernel<<<ROWS, 256, 0, stream>>>(CTX, VG, CTXb);
    convert_bf16_kernel<<<((long)H_DIM * H_DIM + 255) / 256, 256, 0, stream>>>(
        Wo_i, WSTG, (long)H_DIM * H_DIM);
    gemm_kernel<1><<<dim3(H_DIM / 64, ROWS / 128), 256, GEMM_LDS, stream>>>(
        CTXb, WSTG, bo_i, X, X, ROWS, H_DIM, H_DIM);

    // GeGLU FFN
    ln_kernel<<<ROWS, 256, 0, stream>>>(X, nullptr, HS, nullptr, nullptr, H_DIM);
    convert_bf16_kernel<<<((long)I2 * H_DIM + 255) / 256, 256, 0, stream>>>(
        W1_i, WSTG, (long)I2 * H_DIM);
    gemm_kernel<0><<<dim3(I2 / 64, ROWS / 128), 256, GEMM_LDS, stream>>>(
        HS, WSTG, nullptr, nullptr, FF, ROWS, I2, H_DIM);
    geglu_ln_kernel<<<ROWS, 256, 0, stream>>>(FF, FFb);
    convert_bf16_kernel<<<((long)H_DIM * I_DIM + 255) / 256, 256, 0, stream>>>(
        W2_i, WSTG, (long)H_DIM * I_DIM);
    gemm_kernel<1><<<dim3(H_DIM / 64, ROWS / 128), 256, GEMM_LDS, stream>>>(
        FFb, WSTG, nullptr, X, X, ROWS, H_DIM, I_DIM);
  }

  hipMemcpyAsync(d_out, X, (size_t)ROWS * H_DIM * sizeof(float),
                 hipMemcpyDeviceToDevice, stream);
}
